// GAT_9440338117366
// MI455X (gfx1250) — matrix operand order repaired and checked by reading, not compiled, not run on hardware
//
#include <hip/hip_runtime.h>
#include <hip/hip_fp16.h>

#define D   256
#define NH  8
#define CPH 32

typedef __attribute__((ext_vector_type(16))) _Float16 v16h;
typedef __attribute__((ext_vector_type(8)))  _Float16 v8h;
typedef __attribute__((ext_vector_type(8)))  float    v8f;

// ---------- conversions ----------
__global__ void k_cvt_f2h(const float* __restrict__ src, _Float16* __restrict__ dst, int n) {
  int i = blockIdx.x * blockDim.x + threadIdx.x;
  if (i < n) dst[i] = (_Float16)src[i];
}

// W row-major [k][n] (256x256 fp32) -> Wt fp16 [n][k] (transposed, so B-fragment K is contiguous)
__global__ void k_cvt_w_t(const float* __restrict__ W, _Float16* __restrict__ Wt) {
  int i = blockIdx.x * blockDim.x + threadIdx.x;   // 0 .. 65535
  int k = i >> 8, n = i & 255;
  Wt[n * D + k] = (_Float16)W[k * D + n];
}

__global__ void k_fill(float* __restrict__ p, float v, int n) {
  int i = blockIdx.x * blockDim.x + threadIdx.x;
  if (i < n) p[i] = v;
}

// ---------- WMMA GEMM: C[M x 256] = A_h[M x 256] * W  (W given as Wt fp16 [n][k]) ----------
__global__ __launch_bounds__(256)
void k_gemm_wmma(const _Float16* __restrict__ A, const _Float16* __restrict__ Bt,
                 float* __restrict__ Cmat, int M) {
  int wid = blockIdx.x * (blockDim.x >> 5) + (threadIdx.x >> 5);
  int tiles_m = (M + 15) >> 4;
  int tm = wid >> 4;          // m-tile
  int tn = wid & 15;          // 16 n-tiles cover 256 cols
  if (tm >= tiles_m) return;
  int lane = threadIdx.x & 31;
  int hi = lane >> 4;         // half-wave id (0/1)
  int mr = lane & 15;

  int row = tm * 16 + mr; if (row >= M) row = M - 1;
  int col = tn * 16 + mr;
  const _Float16* arow = A  + (size_t)row * D;
  const _Float16* bcol = Bt + (size_t)col * D;

  v8f acc = {};
  #pragma unroll
  for (int k0 = 0; k0 < D; k0 += 32) {
    // A 16x32 f16 layout: lanes 0-15 hold K 0-7 / 16-23; lanes 16-31 hold K 8-15 / 24-31 (same row mr)
    v8h a0 = *(const v8h*)(arow + k0 + hi * 8);
    v8h a1 = *(const v8h*)(arow + k0 + 16 + hi * 8);
    v16h a;
    #pragma unroll
    for (int i = 0; i < 8; ++i) { a[i] = a0[i]; a[i + 8] = a1[i]; }
    // B 32x16 f16 layout: lane n (0-15) = col n with K 0-7 / 16-23; lanes 16-31 = K 8-15 / 24-31
    v8h b0 = *(const v8h*)(bcol + k0 + hi * 8);
    v8h b1 = *(const v8h*)(bcol + k0 + 16 + hi * 8);
    v16h b;
    #pragma unroll
    for (int i = 0; i < 8; ++i) { b[i] = b0[i]; b[i + 8] = b1[i]; }
    acc = __builtin_amdgcn_wmma_f32_16x16x32_f16(false, a, false, b, (short)0, acc, false, false);
  }
  // D layout: VGPR r -> M=r (lanes 0-15) / M=r+8 (lanes 16-31); N = lane&15
  int oc = tn * 16 + mr;
  #pragma unroll
  for (int r = 0; r < 8; ++r) {
    int om = tm * 16 + r + hi * 8;
    if (om < M) Cmat[(size_t)om * D + oc] = acc[r];
  }
}

// ---------- edge phase ----------
__device__ __forceinline__ void atomic_max_f(float* addr, float v) {
  if (v >= 0.0f) atomicMax((int*)addr, __float_as_int(v));
  else           atomicMin((unsigned int*)addr, __float_as_uint(v));
}

__global__ __launch_bounds__(256)
void k_edge_score(const float* __restrict__ xl, const float* __restrict__ xr,
                  const int* __restrict__ esrc, const int* __restrict__ edst,
                  const float* __restrict__ att,
                  float* __restrict__ esc, float* __restrict__ maxb,
                  int Eorig, int Etot) {
  int wid = blockIdx.x * (blockDim.x >> 5) + (threadIdx.x >> 5);
  if (wid >= Etot) return;
  int lane = threadIdx.x & 31;
  int s, d;
  if (wid < Eorig) { s = esrc[wid]; d = edst[wid]; } else { s = wid - Eorig; d = s; }
  const float* pl = xl + (size_t)s * D;
  const float* pr = xr + (size_t)d * D;
  #pragma unroll
  for (int h = 0; h < NH; ++h) {
    float v = pl[h * CPH + lane] + pr[h * CPH + lane];
    v = v > 0.0f ? v : 0.2f * v;                 // LeakyReLU(0.2)
    v *= att[h * CPH + lane];
    #pragma unroll
    for (int off = 16; off > 0; off >>= 1) v += __shfl_xor(v, off, 32);
    if (lane == 0) {
      esc[(size_t)wid * NH + h] = v;
      atomic_max_f(&maxb[d * NH + h], v);
    }
  }
}

__global__ void k_edge_expsum(const int* __restrict__ edst, float* __restrict__ esc,
                              const float* __restrict__ maxb, float* __restrict__ sumb,
                              int Eorig, int Etot) {
  int i = blockIdx.x * blockDim.x + threadIdx.x;
  if (i >= Etot * NH) return;
  int e = i >> 3, h = i & 7;
  int d = (e < Eorig) ? edst[e] : (e - Eorig);
  float ex = __expf(esc[i] - maxb[d * NH + h]);
  esc[i] = ex;
  atomicAdd(&sumb[d * NH + h], ex);
}

__global__ __launch_bounds__(256)
void k_edge_scatter(const float* __restrict__ xl,
                    const int* __restrict__ esrc, const int* __restrict__ edst,
                    const float* __restrict__ esc, const float* __restrict__ sumb,
                    float* __restrict__ acc, int Eorig, int Etot) {
  int wid = blockIdx.x * (blockDim.x >> 5) + (threadIdx.x >> 5);
  if (wid >= Etot) return;
  int lane = threadIdx.x & 31;
  int s, d;
  if (wid < Eorig) { s = esrc[wid]; d = edst[wid]; } else { s = wid - Eorig; d = s; }
  const float* pl = xl + (size_t)s * D;
  float* pa = acc + (size_t)d * D;
  #pragma unroll
  for (int h = 0; h < NH; ++h) {
    float alpha = esc[(size_t)wid * NH + h] / sumb[d * NH + h];
    atomicAdd(&pa[h * CPH + lane], alpha * pl[h * CPH + lane]);
  }
}

// layer1 epilogue: bias + ELU, convert to fp16 for layer2 GEMM input
__global__ void k_finish_elu_cvt(const float* __restrict__ acc, const float* __restrict__ bias,
                                 _Float16* __restrict__ xh, int n) {
  int i = blockIdx.x * blockDim.x + threadIdx.x;
  if (i >= n) return;
  float v = acc[i] + bias[i & (D - 1)];
  v = v > 0.0f ? v : (__expf(v) - 1.0f);         // ELU
  xh[i] = (_Float16)v;
}

__global__ void k_finish_bias(float* __restrict__ out, const float* __restrict__ bias, int n) {
  int i = blockIdx.x * blockDim.x + threadIdx.x;
  if (i < n) out[i] += bias[i & (D - 1)];
}

extern "C" void kernel_launch(void* const* d_in, const int* in_sizes, int n_in,
                              void* d_out, int out_size, void* d_ws, size_t ws_size,
                              hipStream_t stream) {
  const float* x    = (const float*)d_in[0];
  const int*   edg  = (const int*)d_in[1];
  const float* Wl1  = (const float*)d_in[2];
  const float* Wr1  = (const float*)d_in[3];
  const float* att1 = (const float*)d_in[4];
  const float* b1   = (const float*)d_in[5];
  const float* Wl2  = (const float*)d_in[6];
  const float* Wr2  = (const float*)d_in[7];
  const float* att2 = (const float*)d_in[8];
  const float* b2   = (const float*)d_in[9];

  const int N    = in_sizes[0] / D;
  const int E    = in_sizes[1] / 2;
  const int Etot = E + N;                 // with self-loops
  const int* esrc = edg;
  const int* edst = edg + E;

  char* ws = (char*)d_ws;
  size_t off = 0;
  auto alloc = [&](size_t bytes) -> char* {
    char* p = ws + off; off += (bytes + 255) & ~(size_t)255; return p;
  };
  _Float16* xh   = (_Float16*)alloc((size_t)N * D * sizeof(_Float16));
  _Float16* wlt1 = (_Float16*)alloc((size_t)D * D * sizeof(_Float16));
  _Float16* wrt1 = (_Float16*)alloc((size_t)D * D * sizeof(_Float16));
  _Float16* wlt2 = (_Float16*)alloc((size_t)D * D * sizeof(_Float16));
  _Float16* wrt2 = (_Float16*)alloc((size_t)D * D * sizeof(_Float16));
  float* xl   = (float*)alloc((size_t)N * D * sizeof(float));
  float* xr   = (float*)alloc((size_t)N * D * sizeof(float));
  float* esc  = (float*)alloc((size_t)Etot * NH * sizeof(float));
  float* maxb = (float*)alloc((size_t)N * NH * sizeof(float));
  float* sumb = (float*)alloc((size_t)N * NH * sizeof(float));
  float* acc  = (float*)alloc((size_t)N * D * sizeof(float));
  float* outf = (float*)d_out;

  const int nx      = N * D;
  const int nwaves  = ((N + 15) / 16) * 16;      // m-tiles * 16 n-tiles
  const int gblocks = (nwaves + 7) / 8;          // 8 waves / 256-thread block
  const int eblocks = (Etot + 7) / 8;
  const float NEGINF = -__builtin_inff();

  // weight prep (all 4, fp16 transposed)
  k_cvt_w_t<<<(D * D) / 256, 256, 0, stream>>>(Wl1, wlt1);
  k_cvt_w_t<<<(D * D) / 256, 256, 0, stream>>>(Wr1, wrt1);
  k_cvt_w_t<<<(D * D) / 256, 256, 0, stream>>>(Wl2, wlt2);
  k_cvt_w_t<<<(D * D) / 256, 256, 0, stream>>>(Wr2, wrt2);
  k_cvt_f2h<<<(nx + 255) / 256, 256, 0, stream>>>(x, xh, nx);

  // ---- layer 1 ----
  k_gemm_wmma<<<gblocks, 256, 0, stream>>>(xh, wlt1, xl, N);
  k_gemm_wmma<<<gblocks, 256, 0, stream>>>(xh, wrt1, xr, N);
  k_fill<<<(N * NH + 255) / 256, 256, 0, stream>>>(maxb, NEGINF, N * NH);
  k_fill<<<(N * NH + 255) / 256, 256, 0, stream>>>(sumb, 0.0f, N * NH);
  k_fill<<<(nx + 255) / 256, 256, 0, stream>>>(acc, 0.0f, nx);
  k_edge_score<<<eblocks, 256, 0, stream>>>(xl, xr, esrc, edst, att1, esc, maxb, E, Etot);
  k_edge_expsum<<<(Etot * NH + 255) / 256, 256, 0, stream>>>(edst, esc, maxb, sumb, E, Etot);
  k_edge_scatter<<<eblocks, 256, 0, stream>>>(xl, esrc, edst, esc, sumb, acc, E, Etot);
  k_finish_elu_cvt<<<(nx + 255) / 256, 256, 0, stream>>>(acc, b1, xh, nx);

  // ---- layer 2 ----
  k_gemm_wmma<<<gblocks, 256, 0, stream>>>(xh, wlt2, xl, N);
  k_gemm_wmma<<<gblocks, 256, 0, stream>>>(xh, wrt2, xr, N);
  k_fill<<<(N * NH + 255) / 256, 256, 0, stream>>>(maxb, NEGINF, N * NH);
  k_fill<<<(N * NH + 255) / 256, 256, 0, stream>>>(sumb, 0.0f, N * NH);
  k_fill<<<(nx + 255) / 256, 256, 0, stream>>>(outf, 0.0f, nx);
  k_edge_score<<<eblocks, 256, 0, stream>>>(xl, xr, esrc, edst, att2, esc, maxb, E, Etot);
  k_edge_expsum<<<(Etot * NH + 255) / 256, 256, 0, stream>>>(edst, esc, maxb, sumb, E, Etot);
  k_edge_scatter<<<eblocks, 256, 0, stream>>>(xl, esrc, edst, esc, sumb, outf, E, Etot);
  k_finish_bias<<<(nx + 255) / 256, 256, 0, stream>>>(outf, b2, nx);
}
